// SingleHeadAttention_65326452572163
// MI455X (gfx1250) — compile-verified, matrix operand order checked by reading, not run on hardware
//
#include <hip/hip_runtime.h>
#include <hip/hip_bf16.h>

typedef __attribute__((ext_vector_type(16))) __bf16 v16bf;
typedef __attribute__((ext_vector_type(8)))  float  v8f;
typedef __attribute__((ext_vector_type(4)))  int    v4i;

#define D_MODEL  1024
#define SEQ_T    2048
#define NBATCH   4
#define BT_TOTAL 8192

#define K1_ASTR 56     // 32 + 24 pad: 112B row stride (16B aligned), bank-spread
#define K1_BSTR 56
#define QSTR    1048   // 1024 + 24 pad: 2096B stride
#define PSTR    2072   // 2048 + 24 pad: 4144B stride

#ifdef __has_builtin
# if __has_builtin(__builtin_amdgcn_global_load_async_to_lds_b128)
#  define USE_ASYNC_B128 1
# endif
# if __has_builtin(__builtin_amdgcn_s_wait_asynccnt)
#  define HAVE_WAIT_ASYNC 1
# endif
#endif

typedef v4i __attribute__((address_space(1)))* gv4i_p;   // global int4*
typedef v4i __attribute__((address_space(3)))* lv4i_p;   // LDS int4*

union FragBF { uint4 u[2]; v16bf v; };

__device__ __forceinline__ unsigned short f2bf(float f) {
  __bf16 h = (__bf16)f;                     // hardware v_cvt, RNE
  return __builtin_bit_cast(unsigned short, h);
}
__device__ __forceinline__ unsigned packbf2(float lo, float hi) {
  return (unsigned)f2bf(lo) | ((unsigned)f2bf(hi) << 16);
}
__device__ __forceinline__ v8f vzero8() {
  v8f z;
#pragma unroll
  for (int i = 0; i < 8; ++i) z[i] = 0.0f;
  return z;
}
__device__ __forceinline__ FragBF ldfrag(const unsigned short* p) {
  FragBF f;
  f.u[0] = *(const uint4*)p;
  f.u[1] = *(const uint4*)(p + 16);
  return f;
}

// ---------------------------------------------------------------------------
// Kernel 1: P = X @ W  for W in {Wq, Wk, Wv}  (blockIdx.z selects)
//   out bf16; Q pre-scaled by 1/sqrt(d_k); V stored transposed [D][B*T].
// ---------------------------------------------------------------------------
__global__ __launch_bounds__(256) void qkv_gemm_kernel(
    const float* __restrict__ X,  const float* __restrict__ Wq,
    const float* __restrict__ Wk, const float* __restrict__ Wv,
    unsigned short* __restrict__ qo, unsigned short* __restrict__ ko,
    unsigned short* __restrict__ vo)
{
  __shared__ __align__(16) unsigned short As[128 * K1_ASTR];  // X tile, M x K
  __shared__ __align__(16) unsigned short Bs[64 * K1_BSTR];   // W tile, transposed N x K

  const int z = blockIdx.z;
  const float* W       = (z == 0) ? Wq : (z == 1) ? Wk : Wv;
  unsigned short* out  = (z == 0) ? qo : (z == 1) ? ko : vo;
  const float scale    = (z == 0) ? 0.03125f : 1.0f;   // 1/sqrt(1024) folded into Q
  const bool transposed = (z == 2);

  const int mbase = blockIdx.y * 128;
  const int nbase = blockIdx.x * 64;
  const int tid   = threadIdx.x;
  const int wave  = tid >> 5;
  const int lane  = tid & 31;
  const int l16   = lane & 15;
  const int khalf = lane >> 4;

  v8f acc[4];
#pragma unroll
  for (int i = 0; i < 4; ++i) acc[i] = vzero8();

  for (int kc = 0; kc < D_MODEL; kc += 32) {
    __syncthreads();
    // A tile: 128x32 f32 -> bf16.  float4 loads, packed b64 LDS stores.
#pragma unroll
    for (int p = 0; p < 4; ++p) {
      int idx = tid + p * 256;                 // 1024 float4 chunks
      int row = idx >> 3;                      // 8 chunks per row
      int col = (idx & 7) * 4;
      float4 f = *(const float4*)&X[(size_t)(mbase + row) * D_MODEL + kc + col];
      uint2 pk = make_uint2(packbf2(f.x, f.y), packbf2(f.z, f.w));
      *(uint2*)&As[row * K1_ASTR + col] = pk;
    }
    // B tile: 32x64 f32 -> bf16, transposed into LDS (N x K).
#pragma unroll
    for (int p = 0; p < 2; ++p) {
      int idx = tid + p * 256;                 // 512 float4 chunks
      int kr  = idx >> 4;                      // 16 chunks per row
      int col = (idx & 15) * 4;
      float4 f = *(const float4*)&W[(size_t)(kc + kr) * D_MODEL + nbase + col];
      Bs[(col + 0) * K1_BSTR + kr] = f2bf(f.x);
      Bs[(col + 1) * K1_BSTR + kr] = f2bf(f.y);
      Bs[(col + 2) * K1_BSTR + kr] = f2bf(f.z);
      Bs[(col + 3) * K1_BSTR + kr] = f2bf(f.w);
    }
    __syncthreads();

    // Preload A + all four B fragments into distinct registers, then issue
    // the four WMMAs: lets the waits be partial instead of dscnt==0 per op.
    FragBF a = ldfrag(&As[(wave * 16 + l16) * K1_ASTR + khalf * 8]);
    FragBF b0 = ldfrag(&Bs[( 0 + l16) * K1_BSTR + khalf * 8]);
    FragBF b1 = ldfrag(&Bs[(16 + l16) * K1_BSTR + khalf * 8]);
    FragBF b2 = ldfrag(&Bs[(32 + l16) * K1_BSTR + khalf * 8]);
    FragBF b3 = ldfrag(&Bs[(48 + l16) * K1_BSTR + khalf * 8]);
    acc[0] = __builtin_amdgcn_wmma_f32_16x16x32_bf16(false, a.v, false, b0.v, (short)0, acc[0], false, false);
    acc[1] = __builtin_amdgcn_wmma_f32_16x16x32_bf16(false, a.v, false, b1.v, (short)0, acc[1], false, false);
    acc[2] = __builtin_amdgcn_wmma_f32_16x16x32_bf16(false, a.v, false, b2.v, (short)0, acc[2], false, false);
    acc[3] = __builtin_amdgcn_wmma_f32_16x16x32_bf16(false, a.v, false, b3.v, (short)0, acc[3], false, false);
  }

  // Epilogue: branch hoisted outside the store loops (uniform per block).
  const int rowb = mbase + wave * 16 + khalf * 8;
  if (!transposed) {
#pragma unroll
    for (int nt = 0; nt < 4; ++nt) {
      const int col = nbase + nt * 16 + l16;
#pragma unroll
      for (int r = 0; r < 8; ++r)
        out[(size_t)(rowb + r) * D_MODEL + col] = f2bf(acc[nt][r] * scale);
    }
  } else {
#pragma unroll
    for (int nt = 0; nt < 4; ++nt) {
      const int col = nbase + nt * 16 + l16;
#pragma unroll
      for (int r = 0; r < 8; ++r)
        out[(size_t)col * BT_TOTAL + rowb + r] = f2bf(acc[nt][r]);
    }
  }
}

// ---------------------------------------------------------------------------
// Kernel 2: fused causal attention for a 16-query-row block.
// ---------------------------------------------------------------------------
__global__ __launch_bounds__(256) void attn_kernel(
    const unsigned short* __restrict__ qw, const unsigned short* __restrict__ kw,
    const unsigned short* __restrict__ vwT, float* __restrict__ out)
{
  extern __shared__ __align__(16) unsigned short smem[];
  unsigned short* Qs = smem;                   // 16 x QSTR bf16
  unsigned short* Ps = smem + 16 * QSTR;       // 16 x PSTR bf16
  float*          stg = (float*)(Ps + 16 * PSTR);  // [2][8 waves][16 rows]

  const int qb    = blockIdx.x * 16;
  const int b     = blockIdx.y;
  const int tiles = blockIdx.x + 1;            // causal: column tiles 0..tiles-1
  const int tid   = threadIdx.x;
  const int wave  = tid >> 5;
  const int lane  = tid & 31;
  const int l16   = lane & 15;
  const int khalf = lane >> 4;

  // stage Q tile (16 x 1024 bf16) into padded LDS, 16B chunks (async if avail)
  {
    const unsigned short* srow = qw + (size_t)(b * SEQ_T + qb) * D_MODEL;
    for (int i = tid; i < 2048; i += 256) {    // 16 rows x 128 chunks
      int row = i >> 7, c = (i & 127) * 8;
#if USE_ASYNC_B128
      __builtin_amdgcn_global_load_async_to_lds_b128(
          (gv4i_p)(void*)(srow + (size_t)row * D_MODEL + c),
          (lv4i_p)(void*)(Qs + row * QSTR + c), 0, 0);
#else
      *(uint4*)(Qs + row * QSTR + c) =
          *(const uint4*)(srow + (size_t)row * D_MODEL + c);
#endif
    }
#if USE_ASYNC_B128
# if HAVE_WAIT_ASYNC
    __builtin_amdgcn_s_wait_asynccnt(0);
# else
    asm volatile("s_wait_asynccnt 0x0" ::: "memory");
# endif
#endif
  }
  __syncthreads();

  v8f sc[16];
#pragma unroll
  for (int i = 0; i < 16; ++i) sc[i] = vzero8();

  const unsigned short* kbase = kw + (size_t)b * SEQ_T * D_MODEL;

  // S = Q K^T : A-frag from LDS reused across this wave's column tiles
  for (int kc = 0; kc < D_MODEL; kc += 32) {
    FragBF a = ldfrag(Qs + l16 * QSTR + kc + khalf * 8);
#pragma unroll
    for (int lt = 0; lt < 16; ++lt) {
      const int t = wave + (lt << 3);
      if (t < tiles) {                         // block/wave-uniform: EXEC stays full
        FragBF bb = ldfrag(kbase + (size_t)(t * 16 + l16) * D_MODEL + kc + khalf * 8);
        sc[lt] = __builtin_amdgcn_wmma_f32_16x16x32_bf16(
            false, a.v, false, bb.v, (short)0, sc[lt], false, false);
      }
    }
  }

  // causal mask + per-row max
  float rmax[8];
#pragma unroll
  for (int r = 0; r < 8; ++r) rmax[r] = -__builtin_inff();
#pragma unroll
  for (int lt = 0; lt < 16; ++lt) {
    const int t = wave + (lt << 3);
    if (t < tiles) {
      const int col = t * 16 + l16;
#pragma unroll
      for (int r = 0; r < 8; ++r) {
        const int qrow = qb + khalf * 8 + r;
        float v = sc[lt][r];
        if (col > qrow) v = -__builtin_inff();
        sc[lt][r] = v;
        rmax[r] = fmaxf(rmax[r], v);
      }
    }
  }
#pragma unroll
  for (int r = 0; r < 8; ++r)
#pragma unroll
    for (int off = 8; off > 0; off >>= 1)
      rmax[r] = fmaxf(rmax[r], __shfl_xor(rmax[r], off, 32));
  if (l16 == 0)
#pragma unroll
    for (int r = 0; r < 8; ++r) stg[wave * 16 + khalf * 8 + r] = rmax[r];
  __syncthreads();
#pragma unroll
  for (int r = 0; r < 8; ++r) {
    const int row = khalf * 8 + r;
    float m = stg[row];
#pragma unroll
    for (int w = 1; w < 8; ++w) m = fmaxf(m, stg[w * 16 + row]);
    rmax[r] = m;
  }

  // exp + row sum
  float rsum[8];
#pragma unroll
  for (int r = 0; r < 8; ++r) rsum[r] = 0.0f;
#pragma unroll
  for (int lt = 0; lt < 16; ++lt) {
    const int t = wave + (lt << 3);
    if (t < tiles) {
#pragma unroll
      for (int r = 0; r < 8; ++r) {
        float e = __expf(sc[lt][r] - rmax[r]);
        sc[lt][r] = e;
        rsum[r] += e;
      }
    }
  }
#pragma unroll
  for (int r = 0; r < 8; ++r)
#pragma unroll
    for (int off = 8; off > 0; off >>= 1)
      rsum[r] += __shfl_xor(rsum[r], off, 32);
  if (l16 == 0)
#pragma unroll
    for (int r = 0; r < 8; ++r) stg[128 + wave * 16 + khalf * 8 + r] = rsum[r];
  __syncthreads();
#pragma unroll
  for (int r = 0; r < 8; ++r) {
    const int row = khalf * 8 + r;
    float s = 0.0f;
#pragma unroll
    for (int w = 0; w < 8; ++w) s += stg[128 + w * 16 + row];
    rsum[r] = 1.0f / s;
  }

  // normalized P (bf16) -> LDS
#pragma unroll
  for (int lt = 0; lt < 16; ++lt) {
    const int t = wave + (lt << 3);
    if (t < tiles)
#pragma unroll
      for (int r = 0; r < 8; ++r)
        Ps[(khalf * 8 + r) * PSTR + t * 16 + l16] = f2bf(sc[lt][r] * rsum[r]);
  }
  const int smax = tiles * 16;
  const int send = (smax + 31) & ~31;
  if (send != smax) {                          // zero the 16x16 K-pad tile
    int row = tid >> 4, col = smax + (tid & 15);
    Ps[row * PSTR + col] = 0;
  }
  __syncthreads();

  // O = P @ V : each wave owns a 128-wide d-slice; V^T rows are contiguous.
  // B-fragments preloaded in groups of 4 so LDS/VMEM latency overlaps WMMA.
  v8f acc[8];
#pragma unroll
  for (int i = 0; i < 8; ++i) acc[i] = vzero8();
  const int dbase = wave * 128;
  for (int s0 = 0; s0 < send; s0 += 32) {
    FragBF a = ldfrag(Ps + l16 * PSTR + s0 + khalf * 8);
    const unsigned short* vb = vwT + (size_t)b * SEQ_T + s0 + khalf * 8;
#pragma unroll
    for (int g = 0; g < 2; ++g) {
      const int n0 = dbase + g * 64 + l16;
      FragBF f0 = ldfrag(vb + (size_t)(n0 +  0) * BT_TOTAL);
      FragBF f1 = ldfrag(vb + (size_t)(n0 + 16) * BT_TOTAL);
      FragBF f2 = ldfrag(vb + (size_t)(n0 + 32) * BT_TOTAL);
      FragBF f3 = ldfrag(vb + (size_t)(n0 + 48) * BT_TOTAL);
      acc[g*4+0] = __builtin_amdgcn_wmma_f32_16x16x32_bf16(false, a.v, false, f0.v, (short)0, acc[g*4+0], false, false);
      acc[g*4+1] = __builtin_amdgcn_wmma_f32_16x16x32_bf16(false, a.v, false, f1.v, (short)0, acc[g*4+1], false, false);
      acc[g*4+2] = __builtin_amdgcn_wmma_f32_16x16x32_bf16(false, a.v, false, f2.v, (short)0, acc[g*4+2], false, false);
      acc[g*4+3] = __builtin_amdgcn_wmma_f32_16x16x32_bf16(false, a.v, false, f3.v, (short)0, acc[g*4+3], false, false);
    }
  }

#pragma unroll
  for (int nt = 0; nt < 8; ++nt)
#pragma unroll
    for (int r = 0; r < 8; ++r) {
      const int q = qb + khalf * 8 + r;
      const int d = dbase + nt * 16 + l16;
      out[((size_t)(b * SEQ_T + q)) * D_MODEL + d] = acc[nt][r];
    }
}

// ---------------------------------------------------------------------------
extern "C" void kernel_launch(void* const* d_in, const int* in_sizes, int n_in,
                              void* d_out, int out_size, void* d_ws, size_t ws_size,
                              hipStream_t stream) {
  const float* x  = (const float*)d_in[0];
  const float* Wq = (const float*)d_in[1];
  const float* Wk = (const float*)d_in[2];
  const float* Wv = (const float*)d_in[3];

  unsigned short* qws = (unsigned short*)d_ws;                 // Q bf16  [B*T][D]
  unsigned short* kws = qws + (size_t)BT_TOTAL * D_MODEL;      // K bf16  [B*T][D]
  unsigned short* vws = kws + (size_t)BT_TOTAL * D_MODEL;      // V^T bf16 [D][B*T]

  qkv_gemm_kernel<<<dim3(D_MODEL / 64, BT_TOTAL / 128, 3), 256, 0, stream>>>(
      x, Wq, Wk, Wv, qws, kws, vws);

  size_t shmem = (size_t)(16 * QSTR + 16 * PSTR) * sizeof(unsigned short)
               + 256 * sizeof(float);                          // ~101 KB dynamic LDS
  (void)hipFuncSetAttribute((const void*)attn_kernel,
                            hipFuncAttributeMaxDynamicSharedMemorySize, (int)shmem);
  attn_kernel<<<dim3(SEQ_T / 16, NBATCH), 256, shmem, stream>>>(
      qws, kws, vws, (float*)d_out);
}
